// MSRT_38336878084372
// MI455X (gfx1250) — matrix operand order repaired and checked by reading, not compiled, not run on hardware
//
#include <hip/hip_runtime.h>

// ---------------------------------------------------------------------------
// MSRT (chunked retention) forward for MI455X / gfx1250, wave32 + WMMA.
// All matmuls lower to v_wmma_f32_16x16x32_f16 (f16 in, f32 accumulate).
// Projection GEMMs stage the weight tile via the Tensor Data Mover
// (tensor_load_to_lds + s_wait_tensorcnt) when the builtin is available.
// ---------------------------------------------------------------------------

#define B_    2
#define T_    8192
#define HID_  1024
#define NH_   8
#define KD_   128
#define HD_   256
#define CL_   256
#define NC_   32
#define F2_   2048            // HID * FACTOR
#define MT_   (B_ * T_)       // 16384 rows total
#define SCALING_ 0.08838834764831845f   // KD^-0.5

typedef __attribute__((ext_vector_type(16))) _Float16 v16h;
typedef __attribute__((ext_vector_type(8)))  float    v8f;
typedef __attribute__((ext_vector_type(4)))  unsigned int u32x4;
typedef __attribute__((ext_vector_type(8)))  int          i32x8;
typedef __attribute__((ext_vector_type(4)))  int          i32x4;

#if __has_builtin(__builtin_amdgcn_tensor_load_to_lds)
#define USE_TDM 1
#else
#define USE_TDM 0
#endif

typedef __attribute__((address_space(3))) void lds_void;

__device__ __forceinline__ unsigned lds_addr_of(void* p) {
    return (unsigned)(size_t)(lds_void*)p;
}

__device__ __forceinline__ void wait_tensorcnt0() {
#if __has_builtin(__builtin_amdgcn_s_wait_tensorcnt)
    __builtin_amdgcn_s_wait_tensorcnt(0);
#else
    asm volatile("s_wait_tensorcnt 0x0" ::: "memory");
#endif
}

union Frag {
    v16h v;
    uint4 u[2];
    _Float16 h[16];
};

// A fragment (16x32, 16-bit): row = lane&15; lanes 0-15 hold K 0..15,
// lanes 16-31 hold K 16..31.  Two contiguous b128 loads per lane.
__device__ __forceinline__ void load_a_frag(Frag& f, const _Float16* tile,
                                            int lda, int lane) {
    const int row = lane & 15;
    const int kk  = (lane & 16) ? 16 : 0;
    const _Float16* p = tile + (size_t)row * lda + kk;
    f.u[0] = *(const uint4*)(p);
    f.u[1] = *(const uint4*)(p + 8);
}

// B fragment (32x16, 16-bit) from a [N][K] (transposed) tile: col n = lane&15,
// lanes 0-15 hold K 0..15, lanes 16-31 hold K 16..31 (contiguous in K).
__device__ __forceinline__ void load_b_frag(Frag& f, const _Float16* tileT,
                                            int ldb, int lane) {
    const int n  = lane & 15;
    const int kb = (lane & 16) ? 16 : 0;
    const _Float16* p = tileT + (size_t)n * ldb + kb;
    f.u[0] = *(const uint4*)(p);
    f.u[1] = *(const uint4*)(p + 8);
}

__device__ __forceinline__ v8f wmma16(const Frag& a, const Frag& b, v8f c) {
    return __builtin_amdgcn_wmma_f32_16x16x32_f16(
        /*neg_a=*/false, a.v, /*neg_b=*/false, b.v,
        /*c_mod=*/(short)0, c, /*reuse_a=*/false, /*reuse_b=*/false);
}

// ---------------------------------------------------------------------------
// Weight prep: W[K][N] f32  ->  Wt[N][K] f16   (B operand layout)
// ---------------------------------------------------------------------------
__global__ __launch_bounds__(256) void wtrans_kernel(const float* __restrict__ W,
                                                     _Float16* __restrict__ Wt,
                                                     int K, int N) {
    size_t idx = (size_t)blockIdx.x * 256 + threadIdx.x;
    if (idx < (size_t)K * N) {
        int kk = (int)(idx / N);
        int nn = (int)(idx % N);
        Wt[(size_t)nn * K + kk] = (_Float16)W[idx];
    }
}

// ---------------------------------------------------------------------------
// Tiled GEMM:  C[M,N] = (A[M,K] @ Wt^T + bias) * scale   (Wt is [N][K] f16)
// Block tile 128x128, 8 waves (256 threads), K-step 32.
// B tile (f16 weights) is staged into LDS by the Tensor Data Mover with
// a D# descriptor that pads each 32-element row (64B) with 16B -> 40-half
// rows, bank-conflict-free.  A tile is converted f32->f16 by the VALU in
// parallel with the DMA.
// OMODE 0: f32 row-major out.  OMODE 1: f16 row-major + f16 [B][N][T] transposed.
// ---------------------------------------------------------------------------
template <bool AF16, int OMODE>
__global__ __launch_bounds__(256) void gemm_kernel(
    const void* __restrict__ Av, const _Float16* __restrict__ Bt,
    const float* __restrict__ bias, float scale,
    float* __restrict__ outF, _Float16* __restrict__ outH,
    _Float16* __restrict__ outHT,
    int M, int N, int K, int ldT)
{
    __shared__ _Float16 sA[128][40];
    __shared__ _Float16 sB[128][40];

    const int tid  = threadIdx.x;
    const int lane = tid & 31;
    const int wave = tid >> 5;
    const int m0 = blockIdx.x * 128;
    const int n0 = blockIdx.y * 128;
    const int mw = (wave & 3) * 32;   // wave covers 32 rows
    const int nw = (wave >> 2) * 64;  // x 64 cols  (2 m-tiles x 4 n-tiles)

    v8f acc[2][4];
#pragma unroll
    for (int mt = 0; mt < 2; ++mt)
#pragma unroll
        for (int nt = 0; nt < 4; ++nt) acc[mt][nt] = v8f{};

    const int lr = tid >> 1;          // 0..127
    const int lc = (tid & 1) * 16;    // 0 or 16

#if USE_TDM
    const unsigned sB_lds = lds_addr_of(&sB[0][0]);
#endif

    for (int k0 = 0; k0 < K; k0 += 32) {
#if USE_TDM
        // ---- TDM: DMA the 128x32 f16 weight tile into padded LDS rows ----
        if (wave == 0) {
            const unsigned long long ga =
                (unsigned long long)(size_t)(const void*)(Bt + (size_t)n0 * K + k0);
            // D# group0: count=1, lds_addr, global_addr, type=2 ("image")
            u32x4 g0 = { 1u,
                         sB_lds,
                         (unsigned)(ga & 0xFFFFFFFFu),
                         (unsigned)((ga >> 32) & 0x1FFFFFFu) | (2u << 30) };
            // D# group1: data_size=2B, pad_enable, pad every 16 DWORDs by 4 DWORDs
            const unsigned td0 = 0x100000u, td1 = 0x100000u;  // generous OOB dims
            i32x8 g1 = { (int)((1u << 16) | (1u << 20) | (3u << 22) | (3u << 25)),
                         (int)((td0 & 0xFFFFu) << 16),
                         (int)(((td0 >> 16) & 0xFFFFu) | ((td1 & 0xFFFFu) << 16)),
                         (int)(((td1 >> 16) & 0xFFFFu) | (32u << 16)),   // tile_dim0=32
                         (int)(128u),                                    // tile_dim1=128
                         (int)(unsigned)K,                               // dim0 stride
                         0, 0 };
            i32x4 z4 = { 0, 0, 0, 0 };
            i32x8 z8 = { 0, 0, 0, 0, 0, 0, 0, 0 };
            __builtin_amdgcn_tensor_load_to_lds(g0, g1, z4, z4, z8, 0);
        }
#endif
        // ---- stage A tile (convert f32->f16 if needed) ----
        if (AF16) {
            const _Float16* A = (const _Float16*)Av + (size_t)(m0 + lr) * K + k0 + lc;
            *(uint4*)&sA[lr][lc]     = *(const uint4*)(A);
            *(uint4*)&sA[lr][lc + 8] = *(const uint4*)(A + 8);
        } else {
            const float* A = (const float*)Av + (size_t)(m0 + lr) * K + k0 + lc;
            float4 f0 = ((const float4*)A)[0];
            float4 f1 = ((const float4*)A)[1];
            float4 f2 = ((const float4*)A)[2];
            float4 f3 = ((const float4*)A)[3];
            _Float16* dst = &sA[lr][lc];
            dst[0]  = (_Float16)f0.x; dst[1]  = (_Float16)f0.y;
            dst[2]  = (_Float16)f0.z; dst[3]  = (_Float16)f0.w;
            dst[4]  = (_Float16)f1.x; dst[5]  = (_Float16)f1.y;
            dst[6]  = (_Float16)f1.z; dst[7]  = (_Float16)f1.w;
            dst[8]  = (_Float16)f2.x; dst[9]  = (_Float16)f2.y;
            dst[10] = (_Float16)f2.z; dst[11] = (_Float16)f2.w;
            dst[12] = (_Float16)f3.x; dst[13] = (_Float16)f3.y;
            dst[14] = (_Float16)f3.z; dst[15] = (_Float16)f3.w;
        }
#if USE_TDM
        if (wave == 0) wait_tensorcnt0();
#else
        // ---- stage B tile manually ----
        {
            const _Float16* Bp = Bt + (size_t)(n0 + lr) * K + k0 + lc;
            *(uint4*)&sB[lr][lc]     = *(const uint4*)(Bp);
            *(uint4*)&sB[lr][lc + 8] = *(const uint4*)(Bp + 8);
        }
#endif
        __syncthreads();

        // prefetch next A K-slab (emits global_prefetch_b8)
        if (k0 + 32 < K) {
            if (AF16)
                __builtin_prefetch((const _Float16*)Av + (size_t)(m0 + lr) * K + k0 + 32 + lc, 0, 0);
            else
                __builtin_prefetch((const float*)Av + (size_t)(m0 + lr) * K + k0 + 32 + lc, 0, 0);
        }

        Frag a[2], b[4];
#pragma unroll
        for (int mt = 0; mt < 2; ++mt)
            load_a_frag(a[mt], &sA[mw + mt * 16][0], 40, lane);
#pragma unroll
        for (int nt = 0; nt < 4; ++nt)
            load_b_frag(b[nt], &sB[nw + nt * 16][0], 40, lane);
#pragma unroll
        for (int mt = 0; mt < 2; ++mt)
#pragma unroll
            for (int nt = 0; nt < 4; ++nt)
                acc[mt][nt] = wmma16(a[mt], b[nt], acc[mt][nt]);
        __syncthreads();
    }

    // ---- epilogue: D layout is col n = lane&15, rows r + (lane>=16 ? 8 : 0) ----
    const int rofs = (lane & 16) ? 8 : 0;
#pragma unroll
    for (int mt = 0; mt < 2; ++mt) {
#pragma unroll
        for (int nt = 0; nt < 4; ++nt) {
            const int n = n0 + nw + nt * 16 + (lane & 15);
            const float bn = bias[n];
#pragma unroll
            for (int r = 0; r < 8; ++r) {
                const int m = m0 + mw + mt * 16 + r + rofs;
                const float val = (acc[mt][nt][r] + bn) * scale;
                if (OMODE == 0) {
                    outF[(size_t)m * N + n] = val;
                } else {
                    const _Float16 hv = (_Float16)val;
                    outH[(size_t)m * N + n] = hv;
                    const int bb = m >> 13;         // m / T_
                    const int tt = m & (T_ - 1);
                    outHT[((size_t)bb * N + n) * ldT + tt] = hv;
                }
            }
        }
    }
}

// ---------------------------------------------------------------------------
// RoPE: qp,kp (f32, [B*T,1024]) -> qr,kr (f16) and krT (f16, [B][1024][T]).
// k path gets the second SCALING factor.
// ---------------------------------------------------------------------------
__global__ __launch_bounds__(256) void rope_kernel(
    const float* __restrict__ qp, const float* __restrict__ kp,
    const float* __restrict__ sinp, const float* __restrict__ cosp,
    _Float16* __restrict__ qr, _Float16* __restrict__ kr,
    _Float16* __restrict__ krT)
{
    const int bt  = blockIdx.x;
    const int b   = bt >> 13;
    const int t   = bt & (T_ - 1);
    const int tid = threadIdx.x;
    const size_t base = (size_t)bt * HID_;

#pragma unroll
    for (int p = 0; p < 2; ++p) {
        const int ce = tid * 4 + p * 2;     // even column within 1024
        const int d  = ce & (KD_ - 1);      // position within head
        const float se = sinp[(size_t)t * KD_ + d];
        const float so = sinp[(size_t)t * KD_ + d + 1];
        const float cse = cosp[(size_t)t * KD_ + d];
        const float cso = cosp[(size_t)t * KD_ + d + 1];

        const float q1 = qp[base + ce], q2 = qp[base + ce + 1];
        const float qe = q1 * cse - q2 * se;
        const float qo = q2 * cso + q1 * so;
        qr[base + ce]     = (_Float16)qe;
        qr[base + ce + 1] = (_Float16)qo;

        const float k1 = kp[base + ce], k2 = kp[base + ce + 1];
        const float ke = (k1 * cse - k2 * se) * SCALING_;
        const float ko = (k2 * cso + k1 * so) * SCALING_;
        kr[base + ce]     = (_Float16)ke;
        kr[base + ce + 1] = (_Float16)ko;
        krT[((size_t)b * HID_ + ce)     * T_ + t] = (_Float16)ke;
        krT[((size_t)b * HID_ + ce + 1) * T_ + t] = (_Float16)ko;
    }
}

// ---------------------------------------------------------------------------
// qk = (qr @ kr^T) * mask  per (b, chunk, head).  256x256 output, K = 128.
// Writes masked qk (f16) and inner_scale[i] = max(sum_j |qk|, 1).
// ---------------------------------------------------------------------------
__global__ __launch_bounds__(256) void attn_qk_kernel(
    const _Float16* __restrict__ qr, const _Float16* __restrict__ kr,
    const float* __restrict__ mask,
    _Float16* __restrict__ qkout, float* __restrict__ inner_scale)
{
    const int bch = blockIdx.x;
    const int h = bch % NH_;
    const int c = (bch / NH_) % NC_;
    const int b = bch / (NH_ * NC_);
    const int tid  = threadIdx.x;
    const int lane = tid & 31;
    const int wave = tid >> 5;

    __shared__ float rowsum[CL_];
    rowsum[tid] = 0.0f;
    __syncthreads();

    const size_t qbase = ((size_t)b * T_ + (size_t)c * CL_) * HID_ + (size_t)h * KD_;
    const _Float16* Aq = qr + qbase;
    const _Float16* Bk = kr + qbase;
    const float* mh = mask + (size_t)h * CL_ * CL_;
    const size_t obase = (size_t)bch * CL_ * CL_;

    const int mbase = wave * 32;

    Frag afr[2][4];
#pragma unroll
    for (int mt = 0; mt < 2; ++mt)
#pragma unroll
        for (int kq = 0; kq < 4; ++kq)
            load_a_frag(afr[mt][kq], Aq + (size_t)(mbase + mt * 16) * HID_ + kq * 32,
                        HID_, lane);

    const int rofs = (lane & 16) ? 8 : 0;
    for (int nt = 0; nt < 16; ++nt) {
        v8f acc[2]; acc[0] = v8f{}; acc[1] = v8f{};
#pragma unroll
        for (int kq = 0; kq < 4; ++kq) {
            Frag bf;
            load_b_frag(bf, Bk + (size_t)(nt * 16) * HID_ + kq * 32, HID_, lane);
            acc[0] = wmma16(afr[0][kq], bf, acc[0]);
            acc[1] = wmma16(afr[1][kq], bf, acc[1]);
        }
        const int j = nt * 16 + (lane & 15);
#pragma unroll
        for (int mt = 0; mt < 2; ++mt) {
#pragma unroll
            for (int r = 0; r < 8; ++r) {
                const int i = mbase + mt * 16 + r + rofs;
                const float qv = acc[mt][r] * mh[(size_t)i * CL_ + j];
                atomicAdd(&rowsum[i], fabsf(qv));
                qkout[obase + (size_t)i * CL_ + j] = (_Float16)qv;
            }
        }
    }
    __syncthreads();
    inner_scale[(size_t)bch * CL_ + tid] = fmaxf(rowsum[tid], 1.0f);
}

// ---------------------------------------------------------------------------
// inner_raw = qk @ vc  per (b,c,h).  M=N=K=256.  B operand from vpT [B][2048][T].
// ---------------------------------------------------------------------------
__global__ __launch_bounds__(256) void attn_pv_kernel(
    const _Float16* __restrict__ qk, const _Float16* __restrict__ vpT,
    float* __restrict__ inner_out)
{
    const int bch = blockIdx.x;
    const int h = bch % NH_;
    const int c = (bch / NH_) % NC_;
    const int b = bch / (NH_ * NC_);
    const int lane = threadIdx.x & 31;
    const int wave = threadIdx.x >> 5;

    const _Float16* Aq = qk + (size_t)bch * CL_ * CL_;
    const _Float16* Bv = vpT + ((size_t)b * F2_ + (size_t)h * HD_) * T_ + (size_t)c * CL_;
    float* Co = inner_out + (size_t)bch * CL_ * HD_;

    const int mbase = wave * 32;
    const int rofs = (lane & 16) ? 8 : 0;

    for (int nt = 0; nt < 16; ++nt) {
        v8f acc[2]; acc[0] = v8f{}; acc[1] = v8f{};
#pragma unroll
        for (int kq = 0; kq < 8; ++kq) {
            Frag a0, a1, bf;
            load_a_frag(a0, Aq + (size_t)(mbase)      * CL_ + kq * 32, CL_, lane);
            load_a_frag(a1, Aq + (size_t)(mbase + 16) * CL_ + kq * 32, CL_, lane);
            load_b_frag(bf, Bv + (size_t)(nt * 16) * T_ + kq * 32, T_, lane);
            acc[0] = wmma16(a0, bf, acc[0]);
            acc[1] = wmma16(a1, bf, acc[1]);
        }
        const int d = nt * 16 + (lane & 15);
#pragma unroll
        for (int mt = 0; mt < 2; ++mt)
#pragma unroll
            for (int r = 0; r < 8; ++r) {
                const int i = mbase + mt * 16 + r + rofs;
                Co[(size_t)i * HD_ + d] = acc[mt][r];
            }
    }
}

// ---------------------------------------------------------------------------
// kv = kr^T @ (vc * mask[h,-1,j])  per (b,c,h).  M=128(kd), K=256(j), N=256(d).
// A from krT [B][1024][T]; decay row mask[h][CL-1][:] folded into B fragments.
// ---------------------------------------------------------------------------
__global__ __launch_bounds__(256) void attn_kv_kernel(
    const _Float16* __restrict__ krT, const _Float16* __restrict__ vpT,
    const float* __restrict__ mask, float* __restrict__ kvout)
{
    const int bch = blockIdx.x;
    const int h = bch % NH_;
    const int c = (bch / NH_) % NC_;
    const int b = bch / (NH_ * NC_);
    const int tid  = threadIdx.x;
    const int lane = tid & 31;
    const int wave = tid >> 5;

    __shared__ float wrow[CL_];
    wrow[tid] = mask[(size_t)h * CL_ * CL_ + (size_t)(CL_ - 1) * CL_ + tid];
    __syncthreads();

    const _Float16* Ak = krT + ((size_t)b * HID_ + (size_t)h * KD_ + wave * 16) * T_
                             + (size_t)c * CL_;
    const _Float16* Bv = vpT + ((size_t)b * F2_ + (size_t)h * HD_) * T_ + (size_t)c * CL_;
    float* Co = kvout + (size_t)bch * KD_ * HD_;

    Frag afr[8];
#pragma unroll
    for (int kq = 0; kq < 8; ++kq)
        load_a_frag(afr[kq], Ak + kq * 32, T_, lane);

    const int kb = (lane & 16) ? 16 : 0;
    const int rofs = (lane & 16) ? 8 : 0;

    for (int nt = 0; nt < 16; ++nt) {
        v8f acc = v8f{};
#pragma unroll
        for (int kq = 0; kq < 8; ++kq) {
            Frag bf;
            load_b_frag(bf, Bv + (size_t)(nt * 16) * T_ + kq * 32, T_, lane);
            const int jb = kq * 32 + kb;
#pragma unroll
            for (int e = 0; e < 16; ++e)
                bf.h[e] = (_Float16)((float)bf.h[e] * wrow[jb + e]);
            acc = wmma16(afr[kq], bf, acc);
        }
        const int d = nt * 16 + (lane & 15);
#pragma unroll
        for (int r = 0; r < 8; ++r) {
            const int kd = wave * 16 + r + rofs;
            Co[(size_t)kd * HD_ + d] = acc[r];
        }
    }
}

// ---------------------------------------------------------------------------
// Sequential KV scan per (b,h): state[128][256] in registers across 1024 threads.
// Writes kv_rec transposed f16 [bch][d][kd] and cross_scale[bch].
// ---------------------------------------------------------------------------
__global__ __launch_bounds__(1024) void scan_kernel(
    const float* __restrict__ kv, const float* __restrict__ cross_decay,
    _Float16* __restrict__ kvrecT, float* __restrict__ cross_scale)
{
    const int blk = blockIdx.x;          // b*NH + h
    const int b = blk >> 3;
    const int h = blk & 7;
    const int tid = threadIdx.x;
    const int d   = tid & (HD_ - 1);     // 0..255
    const int seg = tid >> 8;            // 0..3 (kd segment of 32)

    const float cd = cross_decay[h];

    __shared__ float red[1024];
    __shared__ float bcast;

    float s[32];
#pragma unroll
    for (int u = 0; u < 32; ++u) s[u] = 0.0f;
    float scale = 1.0f;

    for (int c = 0; c < NC_; ++c) {
        const size_t bch = ((size_t)b * NC_ + c) * NH_ + h;
        // emit kv_rec (old state / old scale), transposed f16
        const float inv = 1.0f / scale;
        _Float16* recp = kvrecT + (bch * HD_ + d) * KD_ + seg * 32;
#pragma unroll
        for (int u = 0; u < 32; ++u) recp[u] = (_Float16)(s[u] * inv);
        if (tid == 0) cross_scale[bch] = scale;

        // state update + abs column sums
        const float* kvp = kv + (bch * KD_ + (size_t)seg * 32) * HD_ + d;
        float part = 0.0f;
#pragma unroll
        for (int u = 0; u < 32; ++u) {
            s[u] = s[u] * cd + kvp[(size_t)u * HD_];
            part += fabsf(s[u]);
        }
        red[tid] = part;
        __syncthreads();
        if (tid < 256)
            red[tid] = red[tid] + red[tid + 256] + red[tid + 512] + red[tid + 768];
        __syncthreads();
        for (int st = 128; st > 0; st >>= 1) {
            if (tid < st) red[tid] = fmaxf(red[tid], red[tid + st]);
            __syncthreads();
        }
        if (tid == 0) bcast = fmaxf(red[0], 1.0f);
        __syncthreads();
        scale = bcast;
        __syncthreads();
    }
}

// ---------------------------------------------------------------------------
// cross_raw = (qr * inner_decay) @ kv_rec  per (b,c,h).  M=256, K=128, N=256.
// Row decay folded into A fragments; B from kvrecT [bch][d][kd] f16.
// ---------------------------------------------------------------------------
__global__ __launch_bounds__(256) void attn_cross_kernel(
    const _Float16* __restrict__ qr, const _Float16* __restrict__ kvrecT,
    const float* __restrict__ inner_decay, float* __restrict__ cross_out)
{
    const int bch = blockIdx.x;
    const int h = bch % NH_;
    const int c = (bch / NH_) % NC_;
    const int b = bch / (NH_ * NC_);
    const int lane = threadIdx.x & 31;
    const int wave = threadIdx.x >> 5;

    const _Float16* Aq = qr + ((size_t)b * T_ + (size_t)c * CL_) * HID_ + (size_t)h * KD_;
    const _Float16* Bk = kvrecT + (size_t)bch * HD_ * KD_;
    float* Co = cross_out + (size_t)bch * CL_ * HD_;

    const int mbase = wave * 32;
    const int rofs = (lane & 16) ? 8 : 0;

    Frag afr[2][4];
#pragma unroll
    for (int mt = 0; mt < 2; ++mt) {
        const int row = mbase + mt * 16 + (lane & 15);
        const float dec = inner_decay[(size_t)h * CL_ + row];
#pragma unroll
        for (int kq = 0; kq < 4; ++kq) {
            load_a_frag(afr[mt][kq],
                        Aq + (size_t)(mbase + mt * 16) * HID_ + kq * 32, HID_, lane);
#pragma unroll
            for (int e = 0; e < 16; ++e)
                afr[mt][kq].h[e] = (_Float16)((float)afr[mt][kq].h[e] * dec);
        }
    }

    for (int nt = 0; nt < 16; ++nt) {
        v8f acc[2]; acc[0] = v8f{}; acc[1] = v8f{};
#pragma unroll
        for (int kq = 0; kq < 4; ++kq) {
            Frag bf;
            load_b_frag(bf, Bk + (size_t)(nt * 16) * KD_ + kq * 32, KD_, lane);
            acc[0] = wmma16(afr[0][kq], bf, acc[0]);
            acc[1] = wmma16(afr[1][kq], bf, acc[1]);
        }
        const int d = nt * 16 + (lane & 15);
#pragma unroll
        for (int mt = 0; mt < 2; ++mt)
#pragma unroll
            for (int r = 0; r < 8; ++r) {
                const int i = mbase + mt * 16 + r + rofs;
                Co[(size_t)i * HD_ + d] = acc[mt][r];
            }
    }
}

// ---------------------------------------------------------------------------
// Combine + per-head layernorm + SiLU gating.  One block per token (b,t).
// X[b*t][2048] f16 = silu(gp) * LN_h( inner/all + cross*cs/all ).
// ---------------------------------------------------------------------------
__global__ __launch_bounds__(256) void combine_kernel(
    const float* __restrict__ inner_raw, const float* __restrict__ cross_raw,
    const float* __restrict__ inner_scale, const float* __restrict__ cross_scale,
    const float* __restrict__ gp, _Float16* __restrict__ X)
{
    const int bt = blockIdx.x;
    const int b = bt >> 13;
    const int t = bt & (T_ - 1);
    const int c = t >> 8;
    const int i = t & (CL_ - 1);
    const int d = threadIdx.x;

    __shared__ float v[F2_];
    __shared__ float mu[NH_], rv[NH_];

#pragma unroll
    for (int h = 0; h < NH_; ++h) {
        const size_t bch = ((size_t)b * NC_ + c) * NH_ + h;
        const float cs  = cross_scale[bch];
        const float isc = inner_scale[bch * CL_ + i];
        const float all = fmaxf(isc, cs);
        const size_t idx = bch * CL_ * HD_ + (size_t)i * HD_ + d;
        v[h * HD_ + d] = inner_raw[idx] / all + cross_raw[idx] * (cs / all);
    }
    __syncthreads();
    if (threadIdx.x < NH_) {
        const int h = threadIdx.x;
        float sum = 0.0f;
        for (int j = 0; j < HD_; ++j) sum += v[h * HD_ + j];
        const float m = sum * (1.0f / HD_);
        float var = 0.0f;
        for (int j = 0; j < HD_; ++j) {
            const float t2 = v[h * HD_ + j] - m;
            var += t2 * t2;
        }
        mu[h] = m;
        rv[h] = rsqrtf(var * (1.0f / HD_) + 1e-6f);
    }
    __syncthreads();
#pragma unroll
    for (int h = 0; h < NH_; ++h) {
        const float g  = gp[(size_t)bt * F2_ + h * HD_ + d];
        const float sg = g / (1.0f + expf(-g));
        const float nn = (v[h * HD_ + d] - mu[h]) * rv[h];
        X[(size_t)bt * F2_ + h * HD_ + d] = (_Float16)(sg * nn);
    }
}

// ---------------------------------------------------------------------------
// Host-side launcher
// ---------------------------------------------------------------------------
extern "C" void kernel_launch(void* const* d_in, const int* in_sizes, int n_in,
                              void* d_out, int out_size, void* d_ws, size_t ws_size,
                              hipStream_t stream) {
    (void)in_sizes; (void)n_in; (void)out_size; (void)ws_size;

    const float* q    = (const float*)d_in[0];
    const float* k    = (const float*)d_in[1];
    const float* v    = (const float*)d_in[2];
    const float* g    = (const float*)d_in[3];
    const float* sinp = (const float*)d_in[4];
    const float* cosp = (const float*)d_in[5];
    const float* mask = (const float*)d_in[6];
    const float* cross_decay = (const float*)d_in[7];
    const float* inner_decay = (const float*)d_in[8];
    const float* Wq = (const float*)d_in[9];   const float* bq = (const float*)d_in[10];
    const float* Wk = (const float*)d_in[11];  const float* bk = (const float*)d_in[12];
    const float* Wv = (const float*)d_in[13];  const float* bv = (const float*)d_in[14];
    const float* Wg = (const float*)d_in[15];  const float* bg = (const float*)d_in[16];
    const float* Wo = (const float*)d_in[17];  const float* bo = (const float*)d_in[18];
    float* out = (float*)d_out;

    char* ws = (char*)d_ws;
    size_t off = 0;
    auto take = [&](size_t bytes) -> char* {
        char* p = ws + off;
        off = (off + bytes + 255) & ~(size_t)255;
        return p;
    };

    _Float16* wqT = (_Float16*)take((size_t)HID_ * HID_ * 2);
    _Float16* wkT = (_Float16*)take((size_t)HID_ * HID_ * 2);
    _Float16* wvT = (_Float16*)take((size_t)HID_ * F2_ * 2);
    _Float16* wgT = (_Float16*)take((size_t)HID_ * F2_ * 2);
    _Float16* woT = (_Float16*)take((size_t)F2_ * HID_ * 2);
    float*    qp  = (float*)take((size_t)MT_ * HID_ * 4);
    float*    kp  = (float*)take((size_t)MT_ * HID_ * 4);
    float*    gp  = (float*)take((size_t)MT_ * F2_ * 4);
    _Float16* vpH = (_Float16*)take((size_t)MT_ * F2_ * 2);
    _Float16* vpT = (_Float16*)take((size_t)MT_ * F2_ * 2);
    _Float16* qr  = (_Float16*)take((size_t)MT_ * HID_ * 2);
    _Float16* kr  = (_Float16*)take((size_t)MT_ * HID_ * 2);
    _Float16* krT = (_Float16*)take((size_t)MT_ * HID_ * 2);
    _Float16* qk  = (_Float16*)take((size_t)B_ * NC_ * NH_ * CL_ * CL_ * 2);
    float*    isc = (float*)take((size_t)B_ * NC_ * NH_ * CL_ * 4);
    float*    inn = (float*)take((size_t)B_ * NC_ * NH_ * CL_ * HD_ * 4);
    float*    kvw = (float*)take((size_t)B_ * NC_ * NH_ * KD_ * HD_ * 4);
    _Float16* kvrT= (_Float16*)take((size_t)B_ * NC_ * NH_ * KD_ * HD_ * 2);
    float*    csc = (float*)take((size_t)B_ * NC_ * NH_ * 4);
    float*    crs = (float*)take((size_t)B_ * NC_ * NH_ * CL_ * HD_ * 4);
    _Float16* X   = (_Float16*)take((size_t)MT_ * F2_ * 2);

    // 1) weight transpose + f16 convert
    wtrans_kernel<<<(HID_ * HID_ + 255) / 256, 256, 0, stream>>>(Wq, wqT, HID_, HID_);
    wtrans_kernel<<<(HID_ * HID_ + 255) / 256, 256, 0, stream>>>(Wk, wkT, HID_, HID_);
    wtrans_kernel<<<(HID_ * F2_ + 255) / 256, 256, 0, stream>>>(Wv, wvT, HID_, F2_);
    wtrans_kernel<<<(HID_ * F2_ + 255) / 256, 256, 0, stream>>>(Wg, wgT, HID_, F2_);
    wtrans_kernel<<<(F2_ * HID_ + 255) / 256, 256, 0, stream>>>(Wo, woT, F2_, HID_);

    // 2) projections
    gemm_kernel<false, 0><<<dim3(MT_ / 128, HID_ / 128), 256, 0, stream>>>(
        q, wqT, bq, 1.0f, qp, nullptr, nullptr, MT_, HID_, HID_, T_);
    gemm_kernel<false, 0><<<dim3(MT_ / 128, HID_ / 128), 256, 0, stream>>>(
        k, wkT, bk, SCALING_, kp, nullptr, nullptr, MT_, HID_, HID_, T_);
    gemm_kernel<false, 1><<<dim3(MT_ / 128, F2_ / 128), 256, 0, stream>>>(
        v, wvT, bv, 1.0f, nullptr, vpH, vpT, MT_, F2_, HID_, T_);
    gemm_kernel<false, 0><<<dim3(MT_ / 128, F2_ / 128), 256, 0, stream>>>(
        g, wgT, bg, 1.0f, gp, nullptr, nullptr, MT_, F2_, HID_, T_);

    // 3) RoPE -> f16 q/k (+ transposed k)
    rope_kernel<<<MT_, 256, 0, stream>>>(qp, kp, sinp, cosp, qr, kr, krT);

    // 4) per-chunk attention pieces
    const int NBCH = B_ * NC_ * NH_;
    attn_qk_kernel<<<NBCH, 256, 0, stream>>>(qr, kr, mask, qk, isc);
    attn_kv_kernel<<<NBCH, 256, 0, stream>>>(krT, vpT, mask, kvw);
    attn_pv_kernel<<<NBCH, 256, 0, stream>>>(qk, vpT, inn);

    // 5) sequential cross-chunk scan
    scan_kernel<<<B_ * NH_, 1024, 0, stream>>>(kvw, cross_decay, kvrT, csc);

    // 6) cross output
    attn_cross_kernel<<<NBCH, 256, 0, stream>>>(qr, kvrT, inner_decay, crs);

    // 7) combine + LN + SiLU gate
    combine_kernel<<<MT_, 256, 0, stream>>>(inn, crs, isc, csc, gp, X);

    // 8) output projection
    gemm_kernel<true, 0><<<dim3(MT_ / 128, HID_ / 128), 256, 0, stream>>>(
        X, woT, bo, 1.0f, out, nullptr, nullptr, MT_, HID_, F2_, T_);
}
